// PacConv2d_601295421590
// MI455X (gfx1250) — matrix operand order repaired and checked by reading, not compiled, not run on hardware
//
#include <hip/hip_runtime.h>
#include <hip/hip_bf16.h>

typedef _Float16 v16h_t __attribute__((ext_vector_type(16)));
typedef _Float16 v8h_t  __attribute__((ext_vector_type(8)));
typedef float    v8f_t  __attribute__((ext_vector_type(8)));
typedef float    v4f_t  __attribute__((ext_vector_type(4)));

#define KS    5
#define PAD   2
#define NB    4
#define CIN   32
#define COUT  32
#define CG    8
#define HH    128
#define WW    128
#define NTAP  (KS*KS)   // 25
#define HW    (HH*WW)

// LDS weight layout: wlds[((t*COUT + o)*CIN + c)]  (c fastest, 64B per (t,o) row)
#define WLDS_ELEMS (NTAP*COUT*CIN)   // 25600 halves = 50 KB
// LDS x tile (f16): [5 rows][128 cols][40 halves] (32 used; pitch 40 halves = 20 dwords
//  -> lane stride 20 dwords mod 64 banks tiles 0,4,...,60: conflict-free b128)
#define XPITCH_H 40
#define XTILE_ELEMS (KS * WW * XPITCH_H)   // 25600 halves = 50 KB

__global__ __launch_bounds__(256)
void pac_wmma_kernel(const float* __restrict__ x,
                     const float* __restrict__ guide,
                     const float* __restrict__ weight,
                     const float* __restrict__ bias,
                     float* __restrict__ out)
{
    __shared__ _Float16 wlds[WLDS_ELEMS];
    __shared__ _Float16 xtile[XTILE_ELEMS];

    const int tid = threadIdx.x;
    const int b   = blockIdx.x / HH;
    const int m   = blockIdx.x % HH;

    const float* gptr = guide + (size_t)b * CG * HW;
    const float* xb   = x     + (size_t)b * CIN * HW;

    // ---- Stage weights into LDS as f16 in B-fragment order [t][o][c] ----
    // global weight layout: weight[o][c][t] flat = (o*CIN + c)*NTAP + t
    for (int i = tid; i < WLDS_ELEMS; i += 256) {
        const int t   = i >> 10;        // / (COUT*CIN)
        const int rem = i & 1023;
        const int o   = rem >> 5;
        const int c   = rem & 31;
        wlds[i] = (_Float16)weight[(o * CIN + c) * NTAP + t];
    }

    // ---- Stage x row-window into LDS as f16, transposed to [row][col][c] ----
    // 5 rows x 32 ch x 32 float4-cols = 5120 float4 loads, 20 per thread, coalesced
    for (int j4 = tid; j4 < KS * CIN * (WW / 4); j4 += 256) {
        const int col4 = (j4 & 31) * 4;          // 0,4,...,124
        const int c    = (j4 >> 5) & 31;         // channel
        const int r    = j4 >> 10;               // window row 0..4
        const int yc   = min(max(m - PAD + r, 0), HH - 1);   // clamped row
        const v4f_t v  = *(const v4f_t*)&xb[(c * HH + yc) * WW + col4];
        xtile[(r * WW + col4 + 0) * XPITCH_H + c] = (_Float16)v.x;
        xtile[(r * WW + col4 + 1) * XPITCH_H + c] = (_Float16)v.y;
        xtile[(r * WW + col4 + 2) * XPITCH_H + c] = (_Float16)v.z;
        xtile[(r * WW + col4 + 3) * XPITCH_H + c] = (_Float16)v.w;
    }
    __syncthreads();

    const int wave   = tid >> 5;        // 0..7
    const int lane   = tid & 31;
    const int l15    = lane & 15;
    const int hiHalf = lane >> 4;       // 0 or 1
    const int n0     = wave * 16;       // M-tile base column
    const int np     = n0 + l15;        // this lane's pixel column (row M=l15)

    // guide center features for this pixel
    float gc[CG];
#pragma unroll
    for (int c = 0; c < CG; ++c)
        gc[c] = gptr[c * HW + m * WW + np];

    v8f_t acc0 = {};   // N-half 0: output channels 0..15
    v8f_t acc1 = {};   // N-half 1: output channels 16..31

    for (int dy = -PAD; dy <= PAD; ++dy) {
        const int  yy  = m + dy;
        const int  yc  = min(max(yy, 0), HH - 1);   // clamped: load always safe
        const bool inY = (yy >= 0) && (yy < HH);
        const int  rr  = dy + PAD;                  // xtile window row

#pragma unroll
        for (int dx = -PAD; dx <= PAD; ++dx) {
            const int  t     = (dy + PAD) * KS + (dx + PAD);
            const int  xx    = np + dx;
            const int  xc    = min(max(xx, 0), WW - 1);
            const bool in    = inY && (xx >= 0) && (xx < WW);
            const int  gbase = yc * WW + xc;        // guide: + c*HW via const offsets

            // ---- adaptive kernel value (f32) ----
            // OOB lanes compute garbage-but-finite s (s>=0 so exp is in (0,1]);
            // the single kv cndmask below zeroes the whole modulated column.
            float s = 0.f;
#pragma unroll
            for (int c = 0; c < CG; ++c) {
                const float gv = gptr[c * HW + gbase];   // unpredicated clamped load
                const float d  = gv - gc[c];
                s += d * d;
            }
            float kv = __expf(-0.5f * s);
            kv = in ? kv : 0.f;                    // one v_cndmask per tap
            const _Float16 kvh = (_Float16)kv;

            // ---- A fragment from f16 LDS x tile (1 conflict-free b128 per group) ----
            // lanes 0-15:  halves 0..7 -> c=0..7,   halves 8..15 -> c=8..15
            // lanes 16-31: halves 0..7 -> c=16..23, halves 8..15 -> c=24..31
            const _Float16* xp = &xtile[(rr * WW + xc) * XPITCH_H + hiHalf * 16];
            const v8h_t x0 = *(const v8h_t*)(xp);        // group 0
            const v8h_t x1 = *(const v8h_t*)(xp + 8);    // group 1
            v16h_t a;
#pragma unroll
            for (int q = 0; q < 8; ++q) {          // -> v_pk_mul_f16 x8
                a[q]     = x0[q] * kvh;
                a[8 + q] = x1[q] * kvh;
            }

            // ---- B fragments from LDS (16 consecutive halves along c) ----
            // lane j (0-15): K = 0..15 (2 per VGPR); lane j+16: K = 16..31
            const _Float16* wb0 = &wlds[((t * COUT) + l15     ) * CIN + hiHalf * 16];
            const _Float16* wb1 = &wlds[((t * COUT) + 16 + l15) * CIN + hiHalf * 16];
            const v16h_t b0 = *(const v16h_t*)wb0;
            const v16h_t b1 = *(const v16h_t*)wb1;

            acc0 = __builtin_amdgcn_wmma_f32_16x16x32_f16(
                /*neg_a=*/false, a, /*neg_b=*/false, b0,
                /*c_mod=*/(short)0, acc0, /*reuse_a=*/false, /*reuse_b=*/false);
            acc1 = __builtin_amdgcn_wmma_f32_16x16x32_f16(
                /*neg_a=*/false, a, /*neg_b=*/false, b1,
                /*c_mod=*/(short)0, acc1, /*reuse_a=*/false, /*reuse_b=*/false);
        }
    }

    // ---- Store D per documented C/D layout + bias ----
    // VGPR r: lanes 0-15 -> (M=r, N=l15); lanes 16-31 -> (M=8+r, N=l15)
    const int o0 = l15;
    const int o1 = 16 + l15;
    const float bias0 = bias[o0];
    const float bias1 = bias[o1];
#pragma unroll
    for (int r = 0; r < 8; ++r) {
        const int M = r + hiHalf * 8;
        const int n = n0 + M;
        out[((b * COUT + o0) * HH + m) * WW + n] = acc0[r] + bias0;
        out[((b * COUT + o1) * HH + m) * WW + n] = acc1[r] + bias1;
    }
}

extern "C" void kernel_launch(void* const* d_in, const int* in_sizes, int n_in,
                              void* d_out, int out_size, void* d_ws, size_t ws_size,
                              hipStream_t stream) {
    const float* x      = (const float*)d_in[0];
    const float* guide  = (const float*)d_in[1];
    const float* weight = (const float*)d_in[2];
    const float* bias   = (const float*)d_in[3];
    float* out = (float*)d_out;

    dim3 grid(NB * HH);   // 512 blocks: one per (batch, row)
    dim3 block(256);      // 8 wave32 waves, one 16-pixel M-tile each
    pac_wmma_kernel<<<grid, block, 0, stream>>>(x, guide, weight, bias, out);
}
